// CompletionNet_43095701848491
// MI455X (gfx1250) — matrix operand order repaired and checked by reading, not compiled, not run on hardware
//
#include <hip/hip_runtime.h>
#include <hip/hip_bf16.h>
#include <stdint.h>

// ---------------------------------------------------------------------------
// CompletionNet encoder head on gfx1250:
//   3x { sparse conv (gather + per-offset GEMM via V_WMMA_F32_16X16X32_BF16)
//        -> deterministic BN stats -> fused BN+ELU }
// C_IN = C0 = C1 = 32 exactly matches the bf16 WMMA K=32 contraction.
// Conv kernel: wave owns 64 rows (4 M-tiles) x 32 cout; weights staged in LDS.
// Invalid neighbors are redirected to a zeroed sentinel row (index N), so the
// gather path needs no per-dword predication and EXEC stays all-ones for WMMA.
// ---------------------------------------------------------------------------

typedef __attribute__((ext_vector_type(16))) __bf16 v16bf;
typedef __attribute__((ext_vector_type(8)))  float  v8f;

static __device__ __forceinline__ uint16_t f2bf(float f) {
    // round-to-nearest-even f32 -> bf16
    uint32_t u = __float_as_uint(f);
    uint32_t r = (u + 0x7FFFu + ((u >> 16) & 1u)) >> 16;
    return (uint16_t)r;
}

// ---------------- f32 -> bf16 convert (+ zero sentinel row) -----------------
__global__ __launch_bounds__(256) void k_f32_to_bf16(
    const float* __restrict__ in, uint16_t* __restrict__ out,
    int totalIn, int totalOut)
{
    int i = blockIdx.x * 256 + threadIdx.x;
    if (i < totalOut) out[i] = (i < totalIn) ? f2bf(in[i]) : (uint16_t)0;
}

// ---------------- weight repack into WMMA B-matrix layout -------------------
// W: [K][32][32] f32 (kin-major, cout stride 1)
// wPack: u32 [K][ntile(2)][lane(32)][j(8)]
//   lane l -> column n = (l&15) + 16*ntile ; K-pair = 16*(l>>4) + 2j, +1
__global__ __launch_bounds__(256) void k_pack_weights(
    const float* __restrict__ W, uint32_t* __restrict__ wPack, int total)
{
    int i = blockIdx.x * 256 + threadIdx.x;
    if (i >= total) return;
    int j     = i & 7;
    int lane  = (i >> 3) & 31;
    int ntile = (i >> 8) & 1;
    int k     = i >> 9;
    int n   = (lane & 15) + ntile * 16;
    int kin = ((lane >> 4) << 4) + 2 * j;
    float f0 = W[(k * 32 + kin) * 32 + n];
    float f1 = W[(k * 32 + kin + 1) * 32 + n];
    wPack[i] = (uint32_t)f2bf(f0) | ((uint32_t)f2bf(f1) << 16);
}

// ---------------- sparse conv: gather + WMMA --------------------------------
// One wave computes 64 voxels x 32 cout (4 M-tiles), accumulating over K
// offsets. Weights live in LDS; each k-offset's B tile feeds 8 WMMAs.
// featsBf has N+1 rows; row N is all zeros (sentinel for idx<0).
__global__ __launch_bounds__(256) void k_spconv_wmma(
    const uint16_t* __restrict__ featsBf,   // (N+1) x 32 bf16
    const int*      __restrict__ nmap,      // K x N
    const uint32_t* __restrict__ wPack,     // K x 2 x 32 x 8
    float*          __restrict__ out,       // N x 32 f32
    int N, int K)
{
    extern __shared__ uint32_t ldsW[];      // K * 512 dwords

    // Stage packed weights into LDS (b128 copies, all 256 threads).
    const int nVec = K * 128;               // uint4 count
    for (int i = threadIdx.x; i < nVec; i += 256)
        ((uint4*)ldsW)[i] = ((const uint4*)wPack)[i];
    __syncthreads();

    const int lane = threadIdx.x & 31;
    const int wave = threadIdx.x >> 5;
    const long long base = ((long long)blockIdx.x * 8 + wave) * 64;
    if (base >= N) return;                  // uniform per wave

    const int m    = lane & 15;             // row within a 16-row M-tile
    const int half = lane >> 4;             // which K-half this lane holds

    v8f acc[4][2] = {};                     // [M-tile][cout-tile]

    union Pun { uint32_t u[8]; v16bf v; };

    for (int k = 0; k < K; ++k) {
        // ---- B tiles from LDS (2 x ds_load_b128 each) ----
        Pun B0, B1;
        {
            const uint4* lb = (const uint4*)(ldsW + (((k * 2 + 0) * 32) + lane) * 8);
            uint4 lo = lb[0], hi = lb[1];
            B0.u[0] = lo.x; B0.u[1] = lo.y; B0.u[2] = lo.z; B0.u[3] = lo.w;
            B0.u[4] = hi.x; B0.u[5] = hi.y; B0.u[6] = hi.z; B0.u[7] = hi.w;
        }
        {
            const uint4* lb = (const uint4*)(ldsW + (((k * 2 + 1) * 32) + lane) * 8);
            uint4 lo = lb[0], hi = lb[1];
            B1.u[0] = lo.x; B1.u[1] = lo.y; B1.u[2] = lo.z; B1.u[3] = lo.w;
            B1.u[4] = hi.x; B1.u[5] = hi.y; B1.u[6] = hi.z; B1.u[7] = hi.w;
        }

        const int* nm = nmap + (long long)k * N;

        #pragma unroll
        for (int t = 0; t < 4; ++t) {
            long long row = base + t * 16 + m;
            // Tail rows clamp the nmap read; their accumulators are never
            // stored, so whatever they gather is harmless.
            int rclamp = row < N ? (int)row : (N - 1);
            int idx  = nm[rclamp];
            int safe = idx < 0 ? N : idx;   // sentinel zero row for invalid

            // A-matrix gather: row 'safe' (32 bf16 = 16 dwords); this lane
            // needs dwords [half*8..+7] -> two b128 loads.
            const uint4* rowPtr = (const uint4*)(featsBf + (long long)safe * 32);
            uint4 lo = rowPtr[2 * half];
            uint4 hi = rowPtr[2 * half + 1];

            Pun A;
            A.u[0] = lo.x; A.u[1] = lo.y; A.u[2] = lo.z; A.u[3] = lo.w;
            A.u[4] = hi.x; A.u[5] = hi.y; A.u[6] = hi.z; A.u[7] = hi.w;

            acc[t][0] = __builtin_amdgcn_wmma_f32_16x16x32_bf16(
                            false, A.v, false, B0.v, (short)0, acc[t][0], false, false);
            acc[t][1] = __builtin_amdgcn_wmma_f32_16x16x32_bf16(
                            false, A.v, false, B1.v, (short)0, acc[t][1], false, false);
        }
    }

    // C/D layout: lanes 0-15 -> (M=r, N=lane); lanes 16-31 -> (M=r+8, N=lane-16)
    #pragma unroll
    for (int t = 0; t < 4; ++t) {
        #pragma unroll
        for (int r = 0; r < 8; ++r) {
            long long row = base + t * 16 + r + half * 8;
            if (row < N) {
                out[row * 32 + m]      = acc[t][0][r];
                out[row * 32 + 16 + m] = acc[t][1][r];
            }
        }
    }
}

// ---------------- BN stats: deterministic two-pass reduction ----------------
// Pass 1: per-block partial sums/sumsqs per channel -> partials[block][64]
__global__ __launch_bounds__(256) void k_bn_stats(
    const float* __restrict__ x, float* __restrict__ partials, int N)
{
    int c    = threadIdx.x & 31;
    int rblk = threadIdx.x >> 5;     // 0..7
    float s = 0.f, s2 = 0.f;
    for (long long r = (long long)blockIdx.x * 8 + rblk; r < N;
         r += (long long)gridDim.x * 8) {
        float v = x[r * 32 + c];
        s += v; s2 += v * v;
    }
    __shared__ float ls[256], ls2[256];
    ls[threadIdx.x] = s; ls2[threadIdx.x] = s2;
    __syncthreads();
    if (rblk == 0) {
        #pragma unroll
        for (int t = 1; t < 8; ++t) { s += ls[t * 32 + c]; s2 += ls2[t * 32 + c]; }
        partials[blockIdx.x * 64 + c]      = s;
        partials[blockIdx.x * 64 + 32 + c] = s2;
    }
}

// Pass 2: fixed-order finalize -> stats[0..31]=sum, stats[32..63]=sumsq
__global__ void k_bn_finalize(const float* __restrict__ partials,
                              float* __restrict__ stats, int nblocks)
{
    int t = threadIdx.x;             // 0..63
    float s = 0.f;
    for (int i = 0; i < nblocks; ++i) s += partials[i * 64 + t];
    stats[t] = s;
}

// ---------------- fused BN + ELU (+ optional bf16 out) ----------------------
// OUT_BF16: writes bf16 and zero-fills the sentinel row (i in [totalIn,totalOut)).
template <bool OUT_BF16>
__global__ __launch_bounds__(256) void k_bn_elu(
    const float* __restrict__ x, const float* __restrict__ stats,
    const float* __restrict__ g, const float* __restrict__ b,
    void* __restrict__ out, int totalIn, int totalOut, float invN)
{
    int i = blockIdx.x * 256 + threadIdx.x;
    if (i >= totalOut) return;
    if (OUT_BF16 && i >= totalIn) {          // sentinel zero row
        ((uint16_t*)out)[i] = 0;
        return;
    }
    int c = i & 31;
    float mean = stats[c] * invN;
    float var  = stats[32 + c] * invN - mean * mean;
    float rs   = rsqrtf(var + 1e-5f);
    float y    = (x[i] - mean) * rs * g[c] + b[c];
    y = y > 0.f ? y : (expf(y) - 1.f);
    if (OUT_BF16) ((uint16_t*)out)[i] = f2bf(y);
    else          ((float*)out)[i]    = y;
}

// ---------------------------------------------------------------------------
extern "C" void kernel_launch(void* const* d_in, const int* in_sizes, int n_in,
                              void* d_out, int out_size, void* d_ws, size_t ws_size,
                              hipStream_t stream)
{
    const float* feats = (const float*)d_in[0];
    const float* W1    = (const float*)d_in[1];
    const float* g1    = (const float*)d_in[2];
    const float* b1    = (const float*)d_in[3];
    const float* W2    = (const float*)d_in[4];
    const float* g2    = (const float*)d_in[5];
    const float* b2    = (const float*)d_in[6];
    const float* W3    = (const float*)d_in[7];
    const float* g3    = (const float*)d_in[8];
    const float* b3    = (const float*)d_in[9];
    const int*   nmap3 = (const int*)d_in[10];
    const int*   nmap2 = (const int*)d_in[11];

    const int N        = in_sizes[0] / 32;
    const int total    = N * 32;
    const int totalPad = total + 32;          // + sentinel zero row
    const float invN   = 1.0f / (float)N;

    // ---- workspace carve-up ----
    uint8_t* ws = (uint8_t*)d_ws;
    size_t o = 0;
    auto carve = [&](size_t bytes) -> void* {
        void* p = ws + o;
        o = (o + bytes + 255) & ~(size_t)255;
        return p;
    };
    uint16_t* bufA   = (uint16_t*)carve((size_t)totalPad * 2); // bf16 ping (+sentinel)
    uint16_t* bufB   = (uint16_t*)carve((size_t)totalPad * 2); // bf16 pong (+sentinel)
    float*    conv   = (float*)   carve((size_t)total * 4);    // conv f32 out
    uint32_t* wp1    = (uint32_t*)carve((size_t)27 * 512 * 4);
    uint32_t* wp2    = (uint32_t*)carve((size_t)8  * 512 * 4);
    uint32_t* wp3    = (uint32_t*)carve((size_t)27 * 512 * 4);
    const int STAT_BLOCKS = 512;
    float*    part   = (float*)   carve((size_t)STAT_BLOCKS * 64 * 4);
    float*    stats  = (float*)   carve(64 * 4);

    const int convBlocks  = (N + 511) / 512;     // 8 waves x 64 rows per block
    const int elemBlocks  = (total + 255) / 256;
    const int elemBlocksP = (totalPad + 255) / 256;
    const size_t lds27 = (size_t)27 * 512 * 4;   // 55296 B
    const size_t lds8  = (size_t)8  * 512 * 4;   // 16384 B

    // feats -> bf16 (+ zero sentinel row)
    k_f32_to_bf16<<<elemBlocksP, 256, 0, stream>>>(feats, bufA, total, totalPad);

    // repack weights into WMMA B layout (bf16 pairs)
    k_pack_weights<<<(27 * 512 + 255) / 256, 256, 0, stream>>>(W1, wp1, 27 * 512);
    k_pack_weights<<<(8  * 512 + 255) / 256, 256, 0, stream>>>(W2, wp2, 8  * 512);
    k_pack_weights<<<(27 * 512 + 255) / 256, 256, 0, stream>>>(W3, wp3, 27 * 512);

    // ---- layer 1: conv k=27 -> BN -> ELU -> bf16 ----
    k_spconv_wmma<<<convBlocks, 256, lds27, stream>>>(bufA, nmap3, wp1, conv, N, 27);
    k_bn_stats<<<STAT_BLOCKS, 256, 0, stream>>>(conv, part, N);
    k_bn_finalize<<<1, 64, 0, stream>>>(part, stats, STAT_BLOCKS);
    k_bn_elu<true><<<elemBlocksP, 256, 0, stream>>>(conv, stats, g1, b1, bufB,
                                                    total, totalPad, invN);

    // ---- layer 2: conv k=8 -> BN -> ELU -> bf16 ----
    k_spconv_wmma<<<convBlocks, 256, lds8, stream>>>(bufB, nmap2, wp2, conv, N, 8);
    k_bn_stats<<<STAT_BLOCKS, 256, 0, stream>>>(conv, part, N);
    k_bn_finalize<<<1, 64, 0, stream>>>(part, stats, STAT_BLOCKS);
    k_bn_elu<true><<<elemBlocksP, 256, 0, stream>>>(conv, stats, g2, b2, bufA,
                                                    total, totalPad, invN);

    // ---- layer 3: conv k=27 -> BN -> ELU -> f32 out ----
    k_spconv_wmma<<<convBlocks, 256, lds27, stream>>>(bufA, nmap3, wp3, conv, N, 27);
    k_bn_stats<<<STAT_BLOCKS, 256, 0, stream>>>(conv, part, N);
    k_bn_finalize<<<1, 64, 0, stream>>>(part, stats, STAT_BLOCKS);
    k_bn_elu<false><<<elemBlocks, 256, 0, stream>>>(conv, stats, g3, b3, d_out,
                                                    total, total, invN);
}
